// SCNNlayer_979252544078
// MI455X (gfx1250) — matrix operand order repaired and checked by reading, not compiled, not run on hardware
//
#include <hip/hip_runtime.h>
#include <stdint.h>

typedef __attribute__((ext_vector_type(16))) __bf16 v16bf;
typedef __attribute__((ext_vector_type(8)))  float  v8f;

#define B_    8
#define T_    30
#define CIN_  64
#define COUT_ 128
#define KDIM  576          // CIN*3*3
#define M_TOT 245760       // B*T*H*W
#define TM    64           // pixels per block tile
#define KC    32           // K-chunk per WMMA step
#define NKC   18           // KDIM/KC
#define AS_STRIDE 40       // halves per LDS A row (32 + pad, 80B keeps 16B align)
#define CB_STRIDE 129      // floats per LDS C row (129 % 64 == 1 -> conflict-free)

__device__ __forceinline__ unsigned short f32_to_bf16(float f) {
  union { float f; unsigned u; } v; v.f = f;
  unsigned u = v.u;
  u += 0x7FFFu + ((u >> 16) & 1u);   // round-to-nearest-even
  return (unsigned short)(u >> 16);
}

// ---- kernel 0: weights f32 -> bf16, wbt[co][K] (K contiguous) --------------
// w_conv is (COUT, CIN, 3, 3) so flat index is already co*576 + (ci*9+kh*3+kw)
__global__ void k_wconv(const float* __restrict__ w, unsigned short* __restrict__ wbt) {
  int i = blockIdx.x * blockDim.x + threadIdx.x;
  if (i < COUT_ * KDIM) wbt[i] = f32_to_bf16(w[i]);
}

// ---- kernel 1: implicit-GEMM conv via bf16 WMMA ----------------------------
__global__ void __launch_bounds__(256)
k_conv(const float* __restrict__ x, const unsigned short* __restrict__ wbt,
       float* __restrict__ y) {
  __shared__ __align__(16) unsigned char smem[TM * CB_STRIDE * 4]; // 33,024 B
  unsigned short* As   = (unsigned short*)smem;   // 64 x 40 halves during K loop
  float*          Cbuf = (float*)smem;            // 64 x 129 floats in epilogue

  const int t    = threadIdx.x;
  const int lane = t & 31;
  const int wv   = t >> 5;
  const int wM   = wv >> 2;          // 0..1 -> 32-row strip of the 64-pixel tile
  const int wN   = wv & 3;           // 0..3 -> 32-col strip of the 128 channels
  const int sel  = lane >> 4;        // half-wave select (WMMA fragment layout)
  const int l16  = lane & 15;

  const int m_base  = blockIdx.x * TM;     // 64 | 1024 -> tile stays in one frame
  const int bt      = m_base >> 10;
  const int hw_base = m_base & 1023;

  v8f acc[2][2] = {};
  union Frag { v16bf v; uint4 q[2]; };

  for (int kc = 0; kc < NKC; ++kc) {
    __syncthreads();
    // im2col gather: 64 pixels x 32 K values -> LDS bf16.
    // Consecutive threads handle consecutive pixels -> contiguous 256B global reads.
    #pragma unroll
    for (int i = 0; i < 8; ++i) {
      int kk  = i * 4 + (t >> 6);          // 0..31
      int row = t & 63;                    // pixel within tile
      int K   = kc * KC + kk;
      int ci  = K / 9;
      int r   = K - ci * 9;
      int r3  = r / 3;
      int dh  = r3 - 1;
      int dw  = (r - r3 * 3) - 1;
      int hw  = hw_base + row;
      int h   = (hw >> 5) + dh;
      int w   = (hw & 31) + dw;
      float v = 0.0f;
      if ((unsigned)h < 32u && (unsigned)w < 32u)
        v = x[(((size_t)bt * CIN_ + ci) << 10) + (h << 5) + w];
      As[row * AS_STRIDE + kk] = f32_to_bf16(v);
    }
    __syncthreads();

    // A fragments from LDS: 16-bit A 16x32 layout:
    // lane<16: M=lane, K 0..7 (V0-3) and 16..23 (V4-7); lane>=16: K 8..15 / 24..31
    Frag a0, a1, b0, b1;
    {
      int r0 = (wM * 32 + l16)      * (AS_STRIDE * 2);
      int r1 = (wM * 32 + 16 + l16) * (AS_STRIDE * 2);
      int o  = sel * 16;
      a0.q[0] = *(const uint4*)(smem + r0 + o);
      a0.q[1] = *(const uint4*)(smem + r0 + 32 + o);
      a1.q[0] = *(const uint4*)(smem + r1 + o);
      a1.q[1] = *(const uint4*)(smem + r1 + 32 + o);
    }
    // B fragments from global (L2-resident 147KB): lane<16 holds K 0..7 / 16..23,
    // lane>=16 holds K 8..15 / 24..31, for column N = base + (lane&15).
    {
      const unsigned short* bp0 =
          wbt + (size_t)(wN * 32 + l16) * KDIM + kc * KC + (sel << 3);
      const unsigned short* bp1 = bp0 + (size_t)16 * KDIM;
      b0.q[0] = *(const uint4*)bp0;  b0.q[1] = *(const uint4*)(bp0 + 16);
      b1.q[0] = *(const uint4*)bp1;  b1.q[1] = *(const uint4*)(bp1 + 16);
    }

    acc[0][0] = __builtin_amdgcn_wmma_f32_16x16x32_bf16(false, a0.v, false, b0.v, (short)0, acc[0][0], false, false);
    acc[0][1] = __builtin_amdgcn_wmma_f32_16x16x32_bf16(false, a0.v, false, b1.v, (short)0, acc[0][1], false, false);
    acc[1][0] = __builtin_amdgcn_wmma_f32_16x16x32_bf16(false, a1.v, false, b0.v, (short)0, acc[1][0], false, false);
    acc[1][1] = __builtin_amdgcn_wmma_f32_16x16x32_bf16(false, a1.v, false, b1.v, (short)0, acc[1][1], false, false);
  }

  // Epilogue: C/D layout (VGPR v: lanes0-15 M=v, lanes16-31 M=8+v; N=lane&15)
  // -> LDS transpose -> coalesced y[bt][co][hw] store.
  __syncthreads();
  #pragma unroll
  for (int mi = 0; mi < 2; ++mi)
    #pragma unroll
    for (int ni = 0; ni < 2; ++ni) {
      int mb = wM * 32 + mi * 16 + (sel << 3);
      int nb = wN * 32 + ni * 16 + l16;
      #pragma unroll
      for (int v = 0; v < 8; ++v)
        Cbuf[(mb + v) * CB_STRIDE + nb] = acc[mi][ni][v];
    }
  __syncthreads();

  const size_t ybase = ((size_t)bt * COUT_) << 10;
  #pragma unroll
  for (int i = 0; i < 32; ++i) {
    int idx = i * 256 + t;        // 0..8191
    int hwl = idx & 63;
    int co  = idx >> 6;
    y[ybase + ((size_t)co << 10) + (hw_base + hwl)] = Cbuf[hwl * CB_STRIDE + co];
  }
}

// ---- kernel 2a: per-channel partial sum/sumsq (deterministic tree) ---------
__global__ void k_reduce_part(const float* __restrict__ y, float* __restrict__ part) {
  int t    = threadIdx.x;
  int co   = t & 127;
  int half = t >> 7;
  float s = 0.f, ss = 0.f;
  const float* base = y + (size_t)blockIdx.x * 512 * COUT_;
  for (int r = half; r < 512; r += 2) {
    float v = base[(size_t)r * COUT_ + co];
    s += v; ss += v * v;
  }
  __shared__ float sh[512];
  sh[t] = s; sh[256 + t] = ss;
  __syncthreads();
  if (half == 0) {
    s  += sh[128 + co];
    ss += sh[256 + 128 + co];
    part[(size_t)blockIdx.x * 256 + co]       = s;
    part[(size_t)blockIdx.x * 256 + 128 + co] = ss;
  }
}

// ---- kernel 2b: finalize mean + gamma*rsqrt(var+eps) -----------------------
__global__ void k_reduce_final(const float* __restrict__ part,
                               const float* __restrict__ gamma,
                               float* __restrict__ stats) {
  int co = threadIdx.x;           // 128 threads
  float s = 0.f, ss = 0.f;
  for (int b = 0; b < 480; ++b) {
    s  += part[(size_t)b * 256 + co];
    ss += part[(size_t)b * 256 + 128 + co];
  }
  const float n = 245760.0f;
  float mean = s / n;
  float var  = ss / n - mean * mean;
  stats[co]       = mean;
  stats[128 + co] = gamma[co] * rsqrtf(var + 1e-5f);
}

// ---- kernel 3: BN + LIF temporal scan (one thread per (b,co,hw)) -----------
__global__ void k_bn_lif(const float* __restrict__ y, const float* __restrict__ stats,
                         const float* __restrict__ beta, const float* __restrict__ alpha,
                         float* __restrict__ out) {
  int tid = blockIdx.x * blockDim.x + threadIdx.x;   // 0..1048575
  int hw  = tid & 1023;
  int co  = (tid >> 10) & 127;
  int b   = tid >> 17;
  float mean  = stats[co];
  float scale = stats[128 + co];
  float bet   = beta[co];
  float al    = alpha[co];
  size_t off = (((size_t)(b * T_) * COUT_ + co) << 10) + hw;
  const float* yp = y + off;
  float*       op = out + off;                        // identical linearization
  float mem = 0.f, sprev = 0.f;
  #pragma unroll
  for (int t = 0; t < T_; ++t) {
    float ybn = (*yp - mean) * scale + bet;
    mem = mem * al + ybn - sprev;                     // VTH * out_prev, VTH=1
    float s = (mem >= 1.0f) ? 1.0f : 0.0f;
    sprev = s;
    *op = s;
    yp += (size_t)COUT_ << 10;
    op += (size_t)COUT_ << 10;
  }
}

extern "C" void kernel_launch(void* const* d_in, const int* in_sizes, int n_in,
                              void* d_out, int out_size, void* d_ws, size_t ws_size,
                              hipStream_t stream) {
  const float* x      = (const float*)d_in[0];
  const float* w_conv = (const float*)d_in[1];
  const float* gamma  = (const float*)d_in[2];
  const float* beta   = (const float*)d_in[3];
  const float* alpha  = (const float*)d_in[4];
  float* out = (float*)d_out;

  unsigned char* ws = (unsigned char*)d_ws;
  float*          y     = (float*)ws;                                   // 125,829,120 B
  unsigned short* wbt   = (unsigned short*)(ws + 125829120u);           //     147,456 B
  float*          part  = (float*)(ws + 125829120u + 147456u);          //     491,520 B
  float*          stats = (float*)(ws + 125829120u + 147456u + 491520u);//       1,024 B

  k_wconv       <<<288,        256, 0, stream>>>(w_conv, wbt);
  k_conv        <<<M_TOT / TM, 256, 0, stream>>>(x, wbt, y);
  k_reduce_part <<<480,        256, 0, stream>>>(y, part);
  k_reduce_final<<<1,          128, 0, stream>>>(part, gamma, stats);
  k_bn_lif      <<<4096,       256, 0, stream>>>(y, stats, beta, alpha, out);
}
